// ResidualAttentionBlock_13838384628323
// MI455X (gfx1250) — matrix operand order repaired and checked by reading, not compiled, not run on hardware
//
#include <hip/hip_runtime.h>
#include <math.h>

// ---------------- types ----------------
typedef __bf16 bf16_t;
typedef __attribute__((ext_vector_type(8)))  bf16_t v8bf;
typedef __attribute__((ext_vector_type(16))) bf16_t v16bf;
typedef __attribute__((ext_vector_type(8)))  float  v8f;
typedef __attribute__((ext_vector_type(4)))  float  f32x4;

static __device__ __forceinline__ v16bf cat16(v8bf lo, v8bf hi) {
    return __builtin_shufflevector(lo, hi, 0,1,2,3,4,5,6,7,8,9,10,11,12,13,14,15);
}

#define WMMA_BF16(a, b, c) \
    __builtin_amdgcn_wmma_f32_16x16x32_bf16(false, (a), false, (b), (short)0, (c), false, false)

// WGP-scope prefetch (default SCOPE=0): pulls into all cache levels incl. WGP$.
// Used in kernel prologues ONLY -- inside hot loops the asm operands break
// address strength-reduction and cause WMMA->VALU hazard v_nops.
#define PREFETCH_WGP(p) \
    asm volatile("global_prefetch_b8 %0, off" :: "v"((const void*)(p)))

// ---------------- constants ----------------
#define BATCH 4
#define SEQ   2048
#define DMODEL 768
#define NHEAD 12
#define HDIM  64
#define DFF   3072
#define ROWS  (BATCH * SEQ)   // 8192

// ---------------- fp32 -> bf16 convert (vectorized x4) ----------------
__global__ void f2bf_kernel(const float* __restrict__ in, bf16_t* __restrict__ out, int n4) {
    int i = blockIdx.x * 256 + threadIdx.x;
    if (i < n4) {
        f32x4 v = ((const f32x4*)in)[i];
        bf16_t* o = out + (size_t)i * 4;
        o[0] = (bf16_t)v.x; o[1] = (bf16_t)v.y; o[2] = (bf16_t)v.z; o[3] = (bf16_t)v.w;
    }
}

// ---------------- LayerNorm: fp32 in -> bf16 out ----------------
__global__ void ln_kernel(const float* __restrict__ x,
                          const float* __restrict__ w,
                          const float* __restrict__ b,
                          bf16_t* __restrict__ out) {
    const int D = DMODEL;
    __shared__ float red[256];
    int row = blockIdx.x, tid = threadIdx.x;
    const float* xr = x + (size_t)row * D;
    float v0 = xr[tid], v1 = xr[tid + 256], v2 = xr[tid + 512];
    red[tid] = v0 + v1 + v2;
    __syncthreads();
    for (int off = 128; off > 0; off >>= 1) {
        if (tid < off) red[tid] += red[tid + off];
        __syncthreads();
    }
    float mean = red[0] * (1.0f / (float)D);
    __syncthreads();
    float d0 = v0 - mean, d1 = v1 - mean, d2 = v2 - mean;
    red[tid] = d0 * d0 + d1 * d1 + d2 * d2;
    __syncthreads();
    for (int off = 128; off > 0; off >>= 1) {
        if (tid < off) red[tid] += red[tid + off];
        __syncthreads();
    }
    float rinv = rsqrtf(red[0] * (1.0f / (float)D) + 1e-5f);
    bf16_t* orow = out + (size_t)row * D;
    orow[tid]       = (bf16_t)(d0 * rinv * w[tid]       + b[tid]);
    orow[tid + 256] = (bf16_t)(d1 * rinv * w[tid + 256] + b[tid + 256]);
    orow[tid + 512] = (bf16_t)(d2 * rinv * w[tid + 512] + b[tid + 512]);
}

// ---------------- generic bf16 WMMA GEMM:  out = A[M,K] * W[N,K]^T + bias ----------------
// wave tile: 32 rows x 64 cols (2 A-fragments x 4 B-fragments -> 8 WMMAs / 32-K step)
// mode 1: bias + exact GELU -> bf16 outb
// mode 2: bias + residual (f32) -> f32 outf
// mode 3: QKV epilogue: Q,K written natural bf16, V written transposed [b,h,d,s]
__global__ void __launch_bounds__(256)
gemm_bf16_kernel(const bf16_t* __restrict__ A,
                 const bf16_t* __restrict__ W,
                 const float*  __restrict__ bias,
                 const float*  __restrict__ resid,
                 float*        __restrict__ outf,
                 bf16_t*       __restrict__ outb,
                 bf16_t*       __restrict__ Qb,
                 bf16_t*       __restrict__ Kb,
                 bf16_t*       __restrict__ Vt,
                 int M, int N, int K, int mode) {
    int tid = threadIdx.x;
    int wave = tid >> 5, lane = tid & 31;
    int lanelo = lane & 15, laneh = lane >> 4;

    size_t m0 = (size_t)blockIdx.x * 256 + wave * 32;
    int n0 = blockIdx.y * 64;

    // A fragment sources: lane holds row (m0 [+16] + lanelo), K halves at laneh*8 and laneh*8+16
    const bf16_t* arow0 = A + (m0 + lanelo) * (size_t)K + laneh * 8;
    const bf16_t* arow1 = arow0 + (size_t)16 * K;
    // B fragment sources: lane holds col (n0+t*16+lanelo), K halves at laneh*8 and laneh*8+16
    const bf16_t* wb[4];
#pragma unroll
    for (int t = 0; t < 4; t++)
        wb[t] = W + (size_t)(n0 + t * 16 + lanelo) * K + laneh * 8;

    // prologue-only WGP-scope prefetch of the first chunk of each stream
    PREFETCH_WGP(arow0);
    PREFETCH_WGP(arow1);
    PREFETCH_WGP(wb[0]);
    PREFETCH_WGP(wb[1]);
    PREFETCH_WGP(wb[2]);
    PREFETCH_WGP(wb[3]);

    v8f acc0[4] = {};
    v8f acc1[4] = {};

    for (int k0 = 0; k0 < K; k0 += 32) {
        v16bf a0 = cat16(*(const v8bf*)(arow0 + k0), *(const v8bf*)(arow0 + k0 + 16));
        v16bf a1 = cat16(*(const v8bf*)(arow1 + k0), *(const v8bf*)(arow1 + k0 + 16));
#pragma unroll
        for (int t = 0; t < 4; t++) {
            v16bf bb = cat16(*(const v8bf*)(wb[t] + k0), *(const v8bf*)(wb[t] + k0 + 16));
            acc0[t] = WMMA_BF16(a0, bb, acc0[t]);
            acc1[t] = WMMA_BF16(a1, bb, acc1[t]);
        }
    }

    // C layout: vgpr r, lane l -> (M = r + (l>>4)*8, N = l&15)
#pragma unroll
    for (int i = 0; i < 2; i++) {
#pragma unroll
        for (int t = 0; t < 4; t++) {
            v8f acc = i ? acc1[t] : acc0[t];
#pragma unroll
            for (int r = 0; r < 8; r++) {
                int row = (int)m0 + i * 16 + r + laneh * 8;
                int col = n0 + t * 16 + lanelo;
                float c = acc[r] + bias[col];
                if (mode == 1) {
                    c = 0.5f * c * (1.0f + erff(c * 0.70710678118654752f));
                    outb[(size_t)row * N + col] = (bf16_t)c;
                } else if (mode == 2) {
                    outf[(size_t)row * N + col] = c + resid[(size_t)row * N + col];
                } else {  // mode 3: QKV scatter
                    if (col < DMODEL) {
                        Qb[(size_t)row * DMODEL + col] = (bf16_t)c;
                    } else if (col < 2 * DMODEL) {
                        Kb[(size_t)row * DMODEL + (col - DMODEL)] = (bf16_t)c;
                    } else {
                        int n2 = col - 2 * DMODEL;
                        int hh = n2 >> 6, dd = n2 & 63;
                        int bb_ = row >> 11, ss = row & (SEQ - 1);
                        Vt[(((size_t)bb_ * NHEAD + hh) * HDIM + dd) * SEQ + ss] = (bf16_t)c;
                    }
                }
            }
        }
    }
}

// ---------------- flash attention: Q,K natural bf16; V transposed; out bf16 natural ----------------
__global__ void __launch_bounds__(256)
attn_kernel(const bf16_t* __restrict__ Qb,
            const bf16_t* __restrict__ Kb,
            const bf16_t* __restrict__ Vt,
            bf16_t* __restrict__ Ob) {
    __shared__ bf16_t plds[8 * 16 * 32];  // per-wave 16x32 bf16 P tile

    int tid = threadIdx.x;
    int wave = tid >> 5, lane = tid & 31;
    int lanelo = lane & 15, laneh = lane >> 4;

    int bh = blockIdx.y;
    int b = bh / NHEAD, h = bh % NHEAD;
    int q0 = blockIdx.x * 128 + wave * 16;

    // ---- Q fragments (held for the whole loop): 16 rows x 64 hd ----
    size_t qrow = (size_t)b * SEQ + q0 + lanelo;
    const bf16_t* qptr = Qb + qrow * DMODEL + h * HDIM + laneh * 8;
    v16bf qa0 = cat16(*(const v8bf*)(qptr),      *(const v8bf*)(qptr + 16));  // hd 0..31
    v16bf qa1 = cat16(*(const v8bf*)(qptr + 32), *(const v8bf*)(qptr + 48));  // hd 32..63

    const bf16_t* kbase = Kb + ((size_t)b * SEQ) * DMODEL + h * HDIM + laneh * 8;
    const bf16_t* vbase = Vt + ((size_t)bh * HDIM) * SEQ;

    // prologue-only WGP-scope prefetch of the first K/V chunk
    PREFETCH_WGP(kbase + (size_t)lanelo * DMODEL);
    PREFETCH_WGP(kbase + (size_t)(16 + lanelo) * DMODEL);
    PREFETCH_WGP(vbase + (size_t)lanelo * SEQ);

    bf16_t* pwave = plds + wave * 512;
    const bf16_t* pw = pwave + lanelo * 32 + laneh * 8;  // A-frag reload address

    float rm[8], rs[8], fac[8];
    v8f o[4] = {};
#pragma unroll
    for (int r = 0; r < 8; r++) { rm[r] = -3.0e38f; rs[r] = 0.0f; }

    for (int kc = 0; kc < SEQ; kc += 32) {
        // ---- scores: two 16x16 key tiles, contraction over hd=64 ----
        v8f s0 = {}, s1 = {};
        const bf16_t* kp0 = kbase + (size_t)(kc + lanelo) * DMODEL;
        const bf16_t* kp1 = kbase + (size_t)(kc + 16 + lanelo) * DMODEL;
        v16bf k00 = cat16(*(const v8bf*)(kp0),      *(const v8bf*)(kp0 + 16));
        v16bf k01 = cat16(*(const v8bf*)(kp0 + 32), *(const v8bf*)(kp0 + 48));
        v16bf k10 = cat16(*(const v8bf*)(kp1),      *(const v8bf*)(kp1 + 16));
        v16bf k11 = cat16(*(const v8bf*)(kp1 + 32), *(const v8bf*)(kp1 + 48));
        s0 = WMMA_BF16(qa0, k00, s0);
        s0 = WMMA_BF16(qa1, k01, s0);
        s1 = WMMA_BF16(qa0, k10, s1);
        s1 = WMMA_BF16(qa1, k11, s1);

        // ---- online softmax (per row; rows striped: vgpr r -> row r + laneh*8) ----
#pragma unroll
        for (int r = 0; r < 8; r++) {
            float a0 = s0[r] * 0.125f;   // 1/sqrt(64)
            float a1 = s1[r] * 0.125f;
            float mx = fmaxf(a0, a1);
            mx = fmaxf(mx, __shfl_xor(mx, 1, 32));
            mx = fmaxf(mx, __shfl_xor(mx, 2, 32));
            mx = fmaxf(mx, __shfl_xor(mx, 4, 32));
            mx = fmaxf(mx, __shfl_xor(mx, 8, 32));
            float mn = fmaxf(rm[r], mx);
            fac[r] = __expf(rm[r] - mn);
            rm[r] = mn;
            float p0 = __expf(a0 - mn);
            float p1 = __expf(a1 - mn);
            float ps = p0 + p1;
            ps += __shfl_xor(ps, 1, 32);
            ps += __shfl_xor(ps, 2, 32);
            ps += __shfl_xor(ps, 4, 32);
            ps += __shfl_xor(ps, 8, 32);
            rs[r] = rs[r] * fac[r] + ps;
            // P tile to LDS, row-major [16][32] bf16
            bf16_t* pr = pwave + (r + laneh * 8) * 32;
            pr[lanelo]      = (bf16_t)p0;
            pr[16 + lanelo] = (bf16_t)p1;
        }
        asm volatile("" ::: "memory");  // keep ds_store -> ds_load order (LDS is in-order per wave)

        // ---- reload P as A fragment (16 rows x 32 keys) ----
        v16bf pa = cat16(*(const v8bf*)(pw), *(const v8bf*)(pw + 16));

        // ---- rescale O accumulators ----
#pragma unroll
        for (int t = 0; t < 4; t++)
#pragma unroll
            for (int r = 0; r < 8; r++) o[t][r] *= fac[r];

        // ---- O += P @ V (V transposed: rows = d, contiguous keys) ----
#pragma unroll
        for (int t = 0; t < 4; t++) {
            const bf16_t* vp = vbase + (size_t)(t * 16 + lanelo) * SEQ + kc + laneh * 8;
            v16bf vb = cat16(*(const v8bf*)(vp), *(const v8bf*)(vp + 16));
            o[t] = WMMA_BF16(pa, vb, o[t]);
        }
    }

    // ---- normalize & write out (bf16, natural [b,s,D] layout) ----
#pragma unroll
    for (int t = 0; t < 4; t++) {
#pragma unroll
        for (int r = 0; r < 8; r++) {
            float val = o[t][r] / rs[r];
            int row = q0 + r + laneh * 8;
            Ob[((size_t)b * SEQ + row) * DMODEL + h * HDIM + t * 16 + lanelo] = (bf16_t)val;
        }
    }
}

// ---------------- launcher ----------------
extern "C" void kernel_launch(void* const* d_in, const int* in_sizes, int n_in,
                              void* d_out, int out_size, void* d_ws, size_t ws_size,
                              hipStream_t stream) {
    const float* x         = (const float*)d_in[0];
    const float* ln1_w     = (const float*)d_in[1];
    const float* ln1_b     = (const float*)d_in[2];
    const float* in_proj_w = (const float*)d_in[3];
    const float* in_proj_b = (const float*)d_in[4];
    const float* out_w     = (const float*)d_in[5];
    const float* out_b     = (const float*)d_in[6];
    const float* ln2_w     = (const float*)d_in[7];
    const float* ln2_b     = (const float*)d_in[8];
    const float* fc_w      = (const float*)d_in[9];
    const float* fc_b      = (const float*)d_in[10];
    const float* proj_w    = (const float*)d_in[11];
    const float* proj_b    = (const float*)d_in[12];

    char* ws = (char*)d_ws;
    // ---- workspace layout (bytes), lifetime-aliased ----
    bf16_t* Wqkv  = (bf16_t*)(ws + 0);                              // 3*768*768*2   = 3,538,944
    bf16_t* Wout  = (bf16_t*)(ws + 3538944);                        // 768*768*2     = 1,179,648
    bf16_t* Wfc   = (bf16_t*)(ws + 4718592);                        // 3072*768*2    = 4,718,592
    bf16_t* Wproj = (bf16_t*)(ws + 9437184);                        // 768*3072*2    = 4,718,592
    bf16_t* hbuf  = (bf16_t*)(ws + 14155776);                       // 8192*768*2    = 12,582,912 (LN1 out, later LN2 out)
    float*  x2    = (float*) (ws + 26738688);                       // 8192*768*4    = 25,165,824
    char*   reg   = ws + 51904512;                                  // 50,331,648 region
    bf16_t* Qb  = (bf16_t*)(reg + 0);
    bf16_t* Kb  = (bf16_t*)(reg + 12582912);
    bf16_t* Vt  = (bf16_t*)(reg + 25165824);
    bf16_t* Ob  = (bf16_t*)(reg + 37748736);
    bf16_t* act = (bf16_t*)(reg + 0);                               // 8192*3072*2 reuses Q/K/Vt/Ob after attention

    float* out = (float*)d_out;

    // 1) weights fp32 -> bf16
    {
        int n4;
        n4 = 3 * DMODEL * DMODEL / 4;
        f2bf_kernel<<<(n4 + 255) / 256, 256, 0, stream>>>(in_proj_w, Wqkv, n4);
        n4 = DMODEL * DMODEL / 4;
        f2bf_kernel<<<(n4 + 255) / 256, 256, 0, stream>>>(out_w, Wout, n4);
        n4 = DFF * DMODEL / 4;
        f2bf_kernel<<<(n4 + 255) / 256, 256, 0, stream>>>(fc_w, Wfc, n4);
        n4 = DMODEL * DFF / 4;
        f2bf_kernel<<<(n4 + 255) / 256, 256, 0, stream>>>(proj_w, Wproj, n4);
    }

    // 2) h = LN1(x)  (bf16)
    ln_kernel<<<ROWS, 256, 0, stream>>>(x, ln1_w, ln1_b, hbuf);

    // 3) QKV = h @ Wqkv^T + b  -> Q,K natural, V transposed
    gemm_bf16_kernel<<<dim3(ROWS / 256, (3 * DMODEL) / 64), 256, 0, stream>>>(
        hbuf, Wqkv, in_proj_b, nullptr, nullptr, nullptr, Qb, Kb, Vt,
        ROWS, 3 * DMODEL, DMODEL, 3);

    // 4) attention -> Ob (bf16)
    attn_kernel<<<dim3(SEQ / 128, BATCH * NHEAD), 256, 0, stream>>>(Qb, Kb, Vt, Ob);

    // 5) x2 = x + Ob @ Wout^T + b  (f32)
    gemm_bf16_kernel<<<dim3(ROWS / 256, DMODEL / 64), 256, 0, stream>>>(
        Ob, Wout, out_b, x, x2, nullptr, nullptr, nullptr, nullptr,
        ROWS, DMODEL, DMODEL, 2);

    // 6) h2 = LN2(x2)  (bf16)
    ln_kernel<<<ROWS, 256, 0, stream>>>(x2, ln2_w, ln2_b, hbuf);

    // 7) act = gelu(h2 @ Wfc^T + fc_b)  (bf16)
    gemm_bf16_kernel<<<dim3(ROWS / 256, DFF / 64), 256, 0, stream>>>(
        hbuf, Wfc, fc_b, nullptr, nullptr, act, nullptr, nullptr, nullptr,
        ROWS, DFF, DMODEL, 1);

    // 8) out = x2 + act @ Wproj^T + proj_b  (f32)
    gemm_bf16_kernel<<<dim3(ROWS / 256, DMODEL / 64), 256, 0, stream>>>(
        act, Wproj, proj_b, x2, out, nullptr, nullptr, nullptr, nullptr,
        ROWS, DMODEL, DFF, 2);
}
